// WaveNet_54589034332776
// MI455X (gfx1250) — matrix operand order repaired and checked by reading, not compiled, not run on hardware
//
#include <hip/hip_runtime.h>

typedef _Float16 h16;
typedef _Float16 v16h __attribute__((ext_vector_type(16)));
typedef _Float16 v8h  __attribute__((ext_vector_type(8)));
typedef float    v8f  __attribute__((ext_vector_type(8)));
typedef float    v4f  __attribute__((ext_vector_type(4)));

#define RCH 64
#define SCH 64
#define NLAYERS 30
#define BB 4
#define TT 32768
#define PAD 512                 // max dilation 2^9
#define HT (PAD + TT)           // padded time extent per batch
#define NT 2                    // 16-col time tiles per wave
#define WAVES 4                 // waves per block

#define HBYTES    ((size_t)BB * HT * RCH * sizeof(h16))
#define SKIPBYTES ((size_t)BB * TT * SCH * sizeof(float))

// dil weights: split-plane A-fragment layout [layer][plane(2)][frag(32)][lane(32)][8h]
#define WDIL_LHALFS (2 * 32 * 32 * 8)            // 16384 halfs per layer (32 KB)
#define WDIL_HALFS  (NLAYERS * WDIL_LHALFS)
#define PLANE_HALFS (32 * 32 * 8)                // 8192 halfs per plane
// rs weights: [layer][frag(16)][lane(32)][16h]
#define WRS_LHALFS  (8 * 2 * 32 * 16)            // 8192 halfs per layer
#define WRS_HALFS   (NLAYERS * WRS_LHALFS)

#define GROW 72                                  // gated slab row stride (halfs), conflict-free
#define GSLAB (16 * GROW)                        // halfs per 16-col tile

// ---------------------------------------------------------------------------
// Pre-pack fp32 weights into fp16 WMMA A-fragment order.
// A 16x32 f16 layout: lane=hl*16+row holds M=16m+row; element e -> K =
// 32*kt + 16*hl + e (contiguous k order, as in the B fragments).
// dil: split planes (p = e/8) so LDS reads are 16B-lane-strided.
// ---------------------------------------------------------------------------
__global__ void prepack_k(const float* __restrict__ dil_w,
                          const float* __restrict__ res_w,
                          const float* __restrict__ skip_w,
                          h16* __restrict__ wdil, h16* __restrict__ wrs) {
    int tid = blockIdx.x * 256 + threadIdx.x;
    if (tid < WDIL_HALFS) {
        int e8 = tid & 7, lane = (tid >> 3) & 31, fr = (tid >> 8) & 31;
        int p = (tid >> 13) & 1, i = tid >> 14;
        int m = fr >> 2, kt = fr & 3;
        int hl = lane >> 4, row = lane & 15;
        int M = 16 * m + row;
        int K = 32 * kt + 16 * hl + 8 * p + e8;
        // K<64: tap 0 applied to h[t-d]; K>=64: tap 1 applied to h[t]
        float v = (K < 64) ? dil_w[((i * 128 + M) * 64 + K) * 2 + 0]
                           : dil_w[((i * 128 + M) * 64 + (K - 64)) * 2 + 1];
        wdil[tid] = (h16)v;
    } else {
        int t2 = tid - WDIL_HALFS;
        if (t2 >= WRS_HALFS) return;
        int e = t2 & 15, lane = (t2 >> 4) & 31, kt = (t2 >> 9) & 1;
        int m = (t2 >> 10) & 7, i = t2 >> 13;
        int hl = lane >> 4, row = lane & 15;
        int M = 16 * m + row;
        int K = 32 * kt + 16 * hl + e;
        float v = (M < 64) ? res_w[(i * 64 + M) * 64 + K]
                           : skip_w[(i * 64 + (M - 64)) * 64 + K];
        wrs[t2] = (h16)v;
    }
}

// ---------------------------------------------------------------------------
// Init: zero pads of both h buffers, zero skip_sum, h0 = start conv (1x1).
// h layout: [b][PAD+t][64ch] fp16 (time-major, channels contiguous).
// ---------------------------------------------------------------------------
__global__ void init_k(const float* __restrict__ x,
                       const float* __restrict__ start_w,
                       const float* __restrict__ start_b,
                       h16* __restrict__ h0, h16* __restrict__ h1,
                       float* __restrict__ skip) {
    int tid = blockIdx.x * 256 + threadIdx.x;       // (b, tt, c)
    int c  = tid & 63;
    int tt = (tid >> 6) % HT;
    int b  = tid / (64 * HT);
    if (b >= BB) return;
    if (tt < PAD) {
        h0[tid] = (h16)0.0f;
        h1[tid] = (h16)0.0f;
    } else {
        int t = tt - PAD;
        h0[tid] = (h16)(start_w[c] * x[(size_t)b * TT + t] + start_b[c]);
        skip[((size_t)b * TT + t) * 64 + c] = 0.0f;
    }
}

// B 32x16 f16 fragment: lane=hl*16+col holds N=col, element e -> K=16*hl+e
// -> 16 contiguous halfs in a [t][ch] layout.
__device__ __forceinline__ v16h load_bfrag(const h16* p) {
    v16h v;
    ((v8h*)&v)[0] = *(const v8h*)p;
    ((v8h*)&v)[1] = *(const v8h*)(p + 8);
    return v;
}

// A fragment from LDS split planes (two 16B-lane-strided ds_load_b128)
__device__ __forceinline__ v16h load_afrag_lds(const h16* lw, int fr, int lane) {
    int fo = (fr * 32 + lane) * 8;
    v16h a;
    ((v8h*)&a)[0] = *(const v8h*)(lw + fo);                // e 0..7
    ((v8h*)&a)[1] = *(const v8h*)(lw + PLANE_HALFS + fo);  // e 8..15
    return a;
}

// A fragment from global packed [frag][lane][16h]
__device__ __forceinline__ v16h load_afrag_g(const h16* w, int fr, int lane) {
    const h16* ap = w + (fr * 32 + lane) * 16;
    v16h a;
    ((v8h*)&a)[0] = *(const v8h*)ap;
    ((v8h*)&a)[1] = *(const v8h*)(ap + 8);
    return a;
}

// ---------------------------------------------------------------------------
// One WaveNet layer. Block = 4 waves; dil weights staged into LDS once.
//   GEMM1: conv[128,t] = Wdil[128x128] x [h(t-d); h(t)]   (4 K-tiles)
//   gated = tanh(filt+bf) * sigmoid(gate+bg)              (via LDS slab)
//   GEMM2: [res;skip][128,t] = Wrs[128x64] x gated        (2 K-tiles)
//   hout = hin + res + rb ;  skip_sum += skip + sb
// A-fragment loops are software-pipelined (prefetch distance 1) so the
// backend can wait on dscnt/loadcnt <= 2 instead of a full drain per WMMA pair.
// ---------------------------------------------------------------------------
__launch_bounds__(WAVES * 32)
__global__ void layer_k(const h16* __restrict__ hin, h16* __restrict__ hout,
                        float* __restrict__ skip,
                        const h16* __restrict__ wdil, const h16* __restrict__ wrs,
                        const float* __restrict__ dil_b,
                        const float* __restrict__ res_b,
                        const float* __restrict__ skip_b,
                        int layer, int dil) {
    __shared__ h16 lds_w[WDIL_LHALFS];              // 32 KB staged dil weights
    __shared__ h16 lds_g[WAVES * NT * GSLAB];       // 18 KB gated slabs (padded rows)

    const int lane = threadIdx.x & 31;
    const int wid  = threadIdx.x >> 5;
    const int hl = lane >> 4, row = lane & 15;

    const int g  = blockIdx.x * WAVES + wid;        // 32-column chunk id
    const int t0 = (g * (16 * NT)) % TT;
    const int b  = (g * (16 * NT)) / TT;
    const h16* hb_in  = hin  + (size_t)b * HT * 64;
    h16*       hb_out = hout + (size_t)b * HT * 64;

    // --- B fragments from h (issue early; overlaps weight staging) ---
    v16h Bf[NT][4];
#pragma unroll
    for (int nt = 0; nt < NT; ++nt) {
        int tbase = PAD + t0 + nt * 16;
#pragma unroll
        for (int kt = 0; kt < 4; ++kt) {
            int tt = (kt < 2) ? (tbase - dil) : tbase;  // pad guarantees >= 0
            int c0 = (kt & 1) * 32;
            Bf[nt][kt] = load_bfrag(hb_in + (size_t)(tt + row) * 64 + c0 + 16 * hl);
        }
    }

    // --- cooperative stage of dil weights: 32 KB, 16B per thread per step ---
    {
        const h16* src = wdil + (size_t)layer * WDIL_LHALFS;
#pragma unroll
        for (int j = 0; j < 16; ++j) {
            int o = (j * 128 + threadIdx.x) * 8;    // 16B chunks, linear
            *(v8h*)(lds_w + o) = *(const v8h*)(src + o);
        }
    }
    __syncthreads();

    // --- GEMM1: 32 fragments (fr = m*4+kt), pipelined A from LDS ---
    v8f acc[8][NT] = {};
    {
        v16h a0 = load_afrag_lds(lds_w, 0, lane);
#pragma unroll
        for (int fr = 0; fr < 32; ++fr) {
            v16h a1 = load_afrag_lds(lds_w, (fr < 31) ? fr + 1 : 31, lane);
            int m = fr >> 2, kt = fr & 3;
#pragma unroll
            for (int nt = 0; nt < NT; ++nt)
                acc[m][nt] = __builtin_amdgcn_wmma_f32_16x16x32_f16(
                    false, a0, false, Bf[nt][kt], (short)0, acc[m][nt], false, false);
            a0 = a1;
        }
    }

    // --- gated nonlinearity -> fp16 slab (row stride 72h: conflict-free) ---
    h16* slab = lds_g + (wid * NT) * GSLAB;
    const float* dbL = dil_b + layer * 128;
#pragma unroll
    for (int m = 0; m < 4; ++m) {
        float bf[8], bg[8];
#pragma unroll
        for (int r = 0; r < 8; ++r) {
            int c = 16 * m + 8 * hl + r;
            bf[r] = dbL[c];
            bg[r] = dbL[64 + c];
        }
#pragma unroll
        for (int nt = 0; nt < NT; ++nt) {
            v8h gv;
#pragma unroll
            for (int r = 0; r < 8; ++r) {
                float f  = acc[m][nt][r] + bf[r];
                float gg = acc[m + 4][nt][r] + bg[r];
                float th = 1.0f - 2.0f / (__expf(2.0f * f) + 1.0f);   // tanh
                float sg = 1.0f / (1.0f + __expf(-gg));               // sigmoid
                gv[r] = (h16)(th * sg);
            }
            *(v8h*)(slab + nt * GSLAB + row * GROW + 16 * m + 8 * hl) = gv;
        }
    }

    // --- reload gated as B fragments (wave-local, DScnt-ordered) ---
    v16h Bg[NT][2];
#pragma unroll
    for (int nt = 0; nt < NT; ++nt)
#pragma unroll
        for (int kt = 0; kt < 2; ++kt)
            Bg[nt][kt] = load_bfrag(slab + nt * GSLAB + row * GROW + kt * 32 + 16 * hl);

    // --- GEMM2: 16 fragments (fr = m*2+kt), pipelined A from global ---
    const h16* wrl = wrs + (size_t)layer * WRS_LHALFS;
    v8f acc2[8][NT] = {};
    {
        v16h a0 = load_afrag_g(wrl, 0, lane);
#pragma unroll
        for (int fr = 0; fr < 16; ++fr) {
            v16h a1 = load_afrag_g(wrl, (fr < 15) ? fr + 1 : 15, lane);
            int m = fr >> 1, kt = fr & 1;
#pragma unroll
            for (int nt = 0; nt < NT; ++nt)
                acc2[m][nt] = __builtin_amdgcn_wmma_f32_16x16x32_f16(
                    false, a0, false, Bg[nt][kt], (short)0, acc2[m][nt], false, false);
            a0 = a1;
        }
    }

    // --- residual: hout = hin + res + rb ---
    const float* rbL = res_b + layer * 64;
    const float* sbL = skip_b + layer * 64;
#pragma unroll
    for (int m = 0; m < 4; ++m) {
        int c0 = 16 * m + 8 * hl;
#pragma unroll
        for (int nt = 0; nt < NT; ++nt) {
            size_t off = (size_t)(PAD + t0 + nt * 16 + row) * 64 + c0;
            v8h hold = *(const v8h*)(hb_in + off);
            v8h hnew;
#pragma unroll
            for (int r = 0; r < 8; ++r)
                hnew[r] = (h16)((float)hold[r] + acc2[m][nt][r] + rbL[c0 + r]);
            *(v8h*)(hb_out + off) = hnew;
        }
    }
    // --- skip accumulate (fp32) ---
    float* sk = skip + (size_t)b * TT * 64;
#pragma unroll
    for (int m = 4; m < 8; ++m) {
        int c0 = 16 * (m - 4) + 8 * hl;
#pragma unroll
        for (int nt = 0; nt < NT; ++nt) {
            float* sp = sk + (size_t)(t0 + nt * 16 + row) * 64 + c0;
            v4f s0 = *(v4f*)sp, s1 = *(v4f*)(sp + 4);
#pragma unroll
            for (int r = 0; r < 4; ++r) {
                s0[r] += acc2[m][nt][r]     + sbL[c0 + r];
                s1[r] += acc2[m][nt][4 + r] + sbL[c0 + 4 + r];
            }
            *(v4f*)sp = s0;
            *(v4f*)(sp + 4) = s1;
        }
    }
}

// ---------------------------------------------------------------------------
// out = end2 @ relu(end1 @ skip_sum + end1_b) + end2_b   (~1 GFLOP, VALU)
// ---------------------------------------------------------------------------
__global__ void end_k(const float* __restrict__ skip,
                      const float* __restrict__ e1w, const float* __restrict__ e1b,
                      const float* __restrict__ e2w, const float* __restrict__ e2b,
                      float* __restrict__ out) {
    __shared__ float w1[64 * 64];
    for (int i = threadIdx.x; i < 64 * 64; i += 256) w1[i] = e1w[i];
    __syncthreads();
    int t = blockIdx.x * 256 + threadIdx.x;     // flattened (b*T + t)
    if (t >= BB * TT) return;
    float s[64];
    const float* sp = skip + (size_t)t * 64;
    for (int c = 0; c < 64; c += 4) {
        v4f v = *(const v4f*)(sp + c);
        s[c] = v[0]; s[c + 1] = v[1]; s[c + 2] = v[2]; s[c + 3] = v[3];
    }
    float o = e2b[0];
    for (int oo = 0; oo < 64; ++oo) {
        float a = e1b[oo];
        for (int c = 0; c < 64; ++c) a = fmaf(w1[oo * 64 + c], s[c], a);
        o = fmaf(e2w[oo], fmaxf(a, 0.0f), o);
    }
    out[t] = o;
}

// ---------------------------------------------------------------------------
extern "C" void kernel_launch(void* const* d_in, const int* in_sizes, int n_in,
                              void* d_out, int out_size, void* d_ws, size_t ws_size,
                              hipStream_t stream) {
    const float* x       = (const float*)d_in[0];
    const float* start_w = (const float*)d_in[1];
    const float* start_b = (const float*)d_in[2];
    const float* dil_w   = (const float*)d_in[3];
    const float* dil_b   = (const float*)d_in[4];
    const float* res_w   = (const float*)d_in[5];
    const float* res_b   = (const float*)d_in[6];
    const float* skip_w  = (const float*)d_in[7];
    const float* skip_b  = (const float*)d_in[8];
    const float* e1w     = (const float*)d_in[9];
    const float* e1b     = (const float*)d_in[10];
    const float* e2w     = (const float*)d_in[11];
    const float* e2b     = (const float*)d_in[12];

    char* ws = (char*)d_ws;
    h16*   h0   = (h16*)ws;
    h16*   h1   = (h16*)(ws + HBYTES);
    float* skip = (float*)(ws + 2 * HBYTES);
    h16*   wdil = (h16*)(ws + 2 * HBYTES + SKIPBYTES);
    h16*   wrs  = wdil + WDIL_HALFS;

    prepack_k<<<(WDIL_HALFS + WRS_HALFS + 255) / 256, 256, 0, stream>>>(
        dil_w, res_w, skip_w, wdil, wrs);
    init_k<<<(BB * HT * 64) / 256, 256, 0, stream>>>(
        x, start_w, start_b, h0, h1, skip);

    h16* hin = h0;
    h16* hout = h1;
    for (int i = 0; i < NLAYERS; ++i) {
        int d = 1 << (i % 10);
        layer_k<<<(BB * TT) / (16 * NT * WAVES), WAVES * 32, 0, stream>>>(
            hin, hout, skip, wdil, wrs, dil_b, res_b, skip_b, i, d);
        h16* tmp = hin; hin = hout; hout = tmp;
    }
    end_k<<<(BB * TT) / 256, 256, 0, stream>>>(
        skip, e1w, e1b, e2w, e2b, (float*)d_out);
}
